// DTransformerBlock_34540126994748
// MI455X (gfx1250) — matrix operand order repaired and checked by reading, not compiled, not run on hardware
//
#include <hip/hip_runtime.h>
#include <hip/hip_bf16.h>
#include <math.h>
#include <stdint.h>

// ---------------------------------------------------------------------------
// Types for gfx1250 WMMA
// ---------------------------------------------------------------------------
typedef __attribute__((ext_vector_type(16))) __bf16       v16bf;
typedef __attribute__((ext_vector_type(8)))  float        v8f;
typedef __attribute__((ext_vector_type(4)))  unsigned int u32x4;

union Frag { v16bf v; u32x4 q[2]; };

static constexpr int L_SEQ = 2048;
static constexpr int DE    = 2048;
static constexpr int DMLP  = 8192;
static constexpr int HEADS = 16;
static constexpr int DH    = 128;
static constexpr int NQKV  = 3 * HEADS * DH;   // 6144
static constexpr int YK    = 160;              // 143 live y columns, padded to 5*32

__device__ __forceinline__ float gelu_tanh(float x) {
    float x3 = x * x * x;
    float t  = tanhf(0.7978845608028654f * (x + 0.044715f * x3));
    return 0.5f * x * (1.0f + t);
}

// CDNA5 async copy: global -> LDS, 16 bytes per lane, tracked by ASYNCcnt.
__device__ __forceinline__ void async_load_b128(unsigned lds_byte_off, uint64_t gaddr) {
    asm volatile("global_load_async_to_lds_b128 %0, %1, off"
                 :: "v"(lds_byte_off), "v"(gaddr)
                 : "memory");
}
__device__ __forceinline__ void wait_async0() {
    asm volatile("s_wait_asynccnt 0x0" ::: "memory");
}
__device__ __forceinline__ unsigned lds_off(const void* p) {
    // generic shared-aperture address: low 32 bits == LDS byte offset
    return (unsigned)(uintptr_t)p;
}

// ---------------------------------------------------------------------------
// Weight conversion / packing kernels (f32 -> bf16)
// ---------------------------------------------------------------------------
__global__ void build_wqkv_kernel(const float* __restrict__ Wq,
                                  const float* __restrict__ Wk,
                                  const float* __restrict__ Wv,
                                  const float* __restrict__ bq,
                                  const float* __restrict__ bk,
                                  const float* __restrict__ bv,
                                  __bf16* __restrict__ Wqkv,
                                  float* __restrict__ bqkv) {
    size_t i = (size_t)blockIdx.x * blockDim.x + threadIdx.x;
    size_t n = (size_t)DE * NQKV;
    if (i >= n) return;
    int d = (int)(i / NQKV);
    int c = (int)(i % NQKV);
    int sel = c / (HEADS * DH);           // 0=q 1=k 2=v
    int hc  = c % (HEADS * DH);
    int h   = hc / DH;
    int e   = hc % DH;
    const float* W = (sel == 0) ? Wq : (sel == 1) ? Wk : Wv;
    Wqkv[i] = (__bf16)W[((size_t)h * DE + d) * DH + e];
    if (d == 0) {
        const float* b = (sel == 0) ? bq : (sel == 1) ? bk : bv;
        bqkv[c] = b[h * DH + e];
    }
}

__global__ void f32_to_bf16_kernel(const float* __restrict__ src,
                                   __bf16* __restrict__ dst, size_t n) {
    size_t i = (size_t)blockIdx.x * blockDim.x + threadIdx.x;
    if (i < n) dst[i] = (__bf16)src[i];
}

__global__ void build_wo_kernel(const float* __restrict__ Wo,
                                __bf16* __restrict__ dst) {
    size_t i = (size_t)blockIdx.x * blockDim.x + threadIdx.x;
    size_t n = (size_t)YK * DE;
    if (i >= n) return;
    int r = (int)(i / DE);
    int c = (int)(i % DE);
    dst[i] = (r < 143) ? (__bf16)Wo[(size_t)r * DE + c] : (__bf16)0.0f;
}

// ---------------------------------------------------------------------------
// Row LayerNorm (population std; std==0 -> 1). 1 workgroup per row.
// ---------------------------------------------------------------------------
__launch_bounds__(256)
__global__ void layernorm_kernel(const float* __restrict__ x,
                                 const float* __restrict__ gamma,
                                 const float* __restrict__ beta,
                                 __bf16* __restrict__ outb,
                                 float* __restrict__ outf) {
    int row  = blockIdx.x;
    int t    = threadIdx.x;
    int lane = t & 31;
    int wid  = t >> 5;
    const float* xr = x + (size_t)row * DE;

    float vals[8];
    float s = 0.f, s2 = 0.f;
#pragma unroll
    for (int i = 0; i < 8; i++) {
        float v = xr[t + 256 * i];
        vals[i] = v;
        s += v;
        s2 += v * v;
    }
#pragma unroll
    for (int off = 16; off >= 1; off >>= 1) {
        s  += __shfl_xor(s, off, 32);
        s2 += __shfl_xor(s2, off, 32);
    }
    __shared__ float red0[8], red1[8];
    if (lane == 0) { red0[wid] = s; red1[wid] = s2; }
    __syncthreads();
    float ts = 0.f, ts2 = 0.f;
#pragma unroll
    for (int w = 0; w < 8; w++) { ts += red0[w]; ts2 += red1[w]; }
    float mean = ts * (1.0f / DE);
    float var  = ts2 * (1.0f / DE) - mean * mean;
    var = fmaxf(var, 0.0f);
    float sd = sqrtf(var);
    if (sd == 0.0f) sd = 1.0f;
    float inv = 1.0f / sd;
#pragma unroll
    for (int i = 0; i < 8; i++) {
        int c = t + 256 * i;
        float o = gamma[c] * ((vals[i] - mean) * inv) + beta[c];
        outb[(size_t)row * DE + c] = (__bf16)o;
        if (outf) outf[(size_t)row * DE + c] = o;
    }
}

// ---------------------------------------------------------------------------
// Generic bf16 WMMA GEMM: C = A(MxK) * B(KxN) + bias, fused epilogues.
//   mode 0: outb = bf16(c)
//   mode 1: outf = 2*aux0 + c           (x2 = 2x + mha)
//   mode 2: outb = bf16(gelu(c))
//   mode 3: outf = aux0 + aux1 + c      (final residual)
// Requires M%128==0, N%128==0, K%32==0.
// 256 threads = 8 waves in 4(m) x 2(n); each wave 32x64 = 2x4 WMMA tiles.
// Double-buffered LDS filled with global_load_async_to_lds_b128 (ASYNCcnt).
// ---------------------------------------------------------------------------
#define GBM 128
#define GBN 128
#define GBK 32
#define LDAS 48    // padded LDS stride (bf16 elems) for A tile
#define LDBS 136   // padded LDS stride (bf16 elems) for B tile

static constexpr int LDS_ABYTES = GBM * LDAS * 2;             // 12288
static constexpr int LDS_BBYTES = GBK * LDBS * 2;             // 8704
static constexpr int LDS_BUF    = LDS_ABYTES + LDS_BBYTES;    // 20992 (16B aligned)

__launch_bounds__(256)
__global__ void gemm_bf16_wmma(const __bf16* __restrict__ A, int lda,
                               const __bf16* __restrict__ B, int ldb,
                               const float* __restrict__ bias,
                               int M, int N, int K, int mode,
                               const float* __restrict__ aux0,
                               const float* __restrict__ aux1,
                               __bf16* __restrict__ outb,
                               float* __restrict__ outf, int ldo) {
    __shared__ __bf16 lds_all[LDS_BUF];   // 2 buffers, LDS_BUF bytes each

    int tid  = threadIdx.x;
    int lane = tid & 31;
    int wave = tid >> 5;
    int wm   = wave >> 1;       // 0..3
    int wn   = wave & 1;        // 0..1
    int ln   = lane & 15;
    int hl   = lane >> 4;
    int c0   = hl ? 8 : 0;

    int bm = blockIdx.y * GBM;
    int bn = blockIdx.x * GBN;

    // --- per-thread staging addresses (one 16B chunk each, x2 for A, x2 for B)
    int ra0 = tid >> 2, ca0 = (tid & 3) * 8;    // A: 4 chunks/row, rows 0..63 (+64)
    int rb0 = tid >> 4, cb0 = (tid & 15) * 8;   // B: 16 chunks/row, rows 0..15 (+16)
    uint64_t gA0 = (uint64_t)(uintptr_t)(A + (size_t)(bm + ra0) * lda + ca0);
    uint64_t gA1 = gA0 + (uint64_t)64 * lda * 2;
    uint64_t gB0 = (uint64_t)(uintptr_t)(B + (size_t)rb0 * ldb + bn + cb0);
    uint64_t gB1 = gB0 + (uint64_t)16 * ldb * 2;
    const uint64_t stepA = (uint64_t)GBK * 2;
    const uint64_t stepB = (uint64_t)GBK * ldb * 2;

    unsigned lA0 = lds_off(&lds_all[ra0 * LDAS + ca0]);
    unsigned lA1 = lA0 + 64 * LDAS * 2;
    unsigned lB0 = lds_off(&lds_all[LDS_ABYTES / 2 + rb0 * LDBS + cb0]);
    unsigned lB1 = lB0 + 16 * LDBS * 2;

    auto issue_tile = [&](int b) {
        unsigned d = b ? (unsigned)LDS_BUF : 0u;
        async_load_b128(lA0 + d, gA0);
        async_load_b128(lA1 + d, gA1);
        async_load_b128(lB0 + d, gB0);
        async_load_b128(lB1 + d, gB1);
        gA0 += stepA; gA1 += stepA; gB0 += stepB; gB1 += stepB;
    };

    v8f acc[2][4] = {};
    const int nk = K / GBK;
    issue_tile(0);

    for (int t = 0, b = 0; t < nk; ++t, b ^= 1) {
        wait_async0();          // my async copies for buffer b are in LDS
        __syncthreads();        // everyone's copies done; prev reads of buf b^1 done
        if (t + 1 < nk) issue_tile(b ^ 1);

        const __bf16* Asb = lds_all + (size_t)b * (LDS_BUF / 2);
        const __bf16* Bsb = Asb + LDS_ABYTES / 2;

        Frag af[2], bfr[4];
#pragma unroll
        for (int mi = 0; mi < 2; mi++) {
            int r = wm * 32 + mi * 16 + ln;
            af[mi].q[0] = *(const u32x4*)(&Asb[r * LDAS + c0]);
            af[mi].q[1] = *(const u32x4*)(&Asb[r * LDAS + c0 + 16]);
        }
#pragma unroll
        for (int ni = 0; ni < 4; ni++) {
            int cb = wn * 64 + ni * 16;
            for (int s = 0; s < 16; s++)
                bfr[ni].v[s] = Bsb[((s & 7) + 8 * hl + (s >> 3) * 16) * LDBS + cb + ln];
        }
#pragma unroll
        for (int mi = 0; mi < 2; mi++)
#pragma unroll
            for (int ni = 0; ni < 4; ni++)
                acc[mi][ni] = __builtin_amdgcn_wmma_f32_16x16x32_bf16(
                    false, af[mi].v, false, bfr[ni].v,
                    (short)0, acc[mi][ni], false, false);
    }

    // --- epilogue ---
#pragma unroll
    for (int mi = 0; mi < 2; mi++) {
#pragma unroll
        for (int ni = 0; ni < 4; ni++) {
#pragma unroll
            for (int r = 0; r < 8; r++) {
                int m = bm + wm * 32 + mi * 16 + r + 8 * hl;
                int n = bn + wn * 64 + ni * 16 + ln;
                float c = acc[mi][ni][r] + bias[n];
                size_t idx = (size_t)m * ldo + n;
                if (mode == 0) {
                    outb[idx] = (__bf16)c;
                } else if (mode == 1) {
                    outf[idx] = 2.0f * aux0[idx] + c;
                } else if (mode == 2) {
                    outb[idx] = (__bf16)gelu_tanh(c);
                } else {
                    outf[idx] = aux0[idx] + aux1[idx] + c;
                }
            }
        }
    }
}

// ---------------------------------------------------------------------------
// Flash attention over QKV buffer (L x 6144, bf16): per (head, 64-row qblock).
// Computes S^T = K*Q^T so softmax stats reduce per-lane down VGPRs.
// 4 waves x 32 threads; each wave owns a 16-row q strip, full D_V=128 output.
// ---------------------------------------------------------------------------
__launch_bounds__(128)
__global__ void flash_attn_kernel(const __bf16* __restrict__ QKV,
                                  float* __restrict__ Aout) {
    const int ldq = NQKV;
    int h    = blockIdx.y;
    int qb   = blockIdx.x;             // 32 blocks of 64 rows
    int tid  = threadIdx.x;
    int lane = tid & 31;
    int w    = tid >> 5;
    int ln   = lane & 15;
    int hl   = lane >> 4;
    int q0   = qb * 64 + w * 16;

    const __bf16* Qp = QKV + h * DH;
    const __bf16* Kp = QKV + HEADS * DH + h * DH;
    const __bf16* Vp = QKV + 2 * HEADS * DH + h * DH;

    __shared__ __bf16 Qt[4][128][24];   // per-wave Q^T (d x q), padded stride
    __shared__ __bf16 Pt[4][16][48];    // per-wave P (q x 32 keys), padded

    // stage Q^T for this wave's 16 q rows
    {
        int q = q0 + ln;
#pragma unroll
        for (int it = 0; it < 8; it++) {
            int d0 = it * 16 + hl * 8;
            u32x4 tq = *(const u32x4*)(&Qp[(size_t)q * ldq + d0]);
            const __bf16* tb = (const __bf16*)&tq;
#pragma unroll
            for (int j = 0; j < 8; j++) Qt[w][d0 + j][ln] = tb[j];
        }
    }
    __syncthreads();

    // resident B-frags of Q^T (32d x 16q each, dk = 0..3)
    Frag qtb[4];
#pragma unroll
    for (int dk = 0; dk < 4; dk++) {
        for (int s = 0; s < 16; s++)
            qtb[dk].v[s] = Qt[w][dk * 32 + (s & 7) + 8 * hl + (s >> 3) * 16][ln];
    }

    v8f O[8] = {};
    float m_i = -1e30f, l_i = 0.0f;
    const float rs = 0.08838834764831845f;   // 1/sqrt(128)

    // strength-reduced per-lane base pointers (advance by 32 rows per iter)
    const __bf16* kp0 = Kp + (size_t)ln * ldq + (hl ? 8 : 0);
    const __bf16* kp1 = kp0 + (size_t)16 * ldq;
    const __bf16* vp  = Vp + ln;
    const size_t jstep = (size_t)32 * ldq;

    for (int jt = 0; jt < L_SEQ / 32; jt++) {
        // S^T tiles: (key16 x q16), 2 key sub-tiles, contract d over 128
        v8f s[2] = {};
#pragma unroll
        for (int sub = 0; sub < 2; sub++) {
            const __bf16* kb = sub ? kp1 : kp0;
#pragma unroll
            for (int dk = 0; dk < 4; dk++) {
                Frag ka;
                const __bf16* p = kb + dk * 32;
                ka.q[0] = *(const u32x4*)p;
                ka.q[1] = *(const u32x4*)(p + 16);
                s[sub] = __builtin_amdgcn_wmma_f32_16x16x32_bf16(
                    false, ka.v, false, qtb[dk].v, (short)0, s[sub], false, false);
            }
        }
        // column (=q) softmax stats; this lane holds column q=ln
        float mloc = -1e30f;
#pragma unroll
        for (int sub = 0; sub < 2; sub++)
#pragma unroll
            for (int r = 0; r < 8; r++) mloc = fmaxf(mloc, s[sub][r]);
        mloc *= rs;
        mloc = fmaxf(mloc, __shfl_xor(mloc, 16, 32));
        float mnew  = fmaxf(m_i, mloc);
        float alpha = __expf(m_i - mnew);
        float psum  = 0.0f;
        alignas(16) __bf16 pb[2][8];
#pragma unroll
        for (int sub = 0; sub < 2; sub++)
#pragma unroll
            for (int r = 0; r < 8; r++) {
                float p = __expf(s[sub][r] * rs - mnew);
                psum += p;
                pb[sub][r] = (__bf16)p;
            }
        psum += __shfl_xor(psum, 16, 32);
        l_i = l_i * alpha + psum;
        m_i = mnew;

        // rescale running O: row q = r + 8*hl, factor from lane (r+8*hl) of half
        float af8[8];
#pragma unroll
        for (int r = 0; r < 8; r++) af8[r] = __shfl(alpha, r + hl * 8, 16);
#pragma unroll
        for (int t = 0; t < 8; t++)
#pragma unroll
            for (int r = 0; r < 8; r++) O[t][r] *= af8[r];

        // write P (q x 32key) through LDS to get A-layout frags
#pragma unroll
        for (int sub = 0; sub < 2; sub++)
            *(u32x4*)(&Pt[w][ln][hl * 8 + sub * 16]) = *(const u32x4*)pb[sub];

        Frag pa;
        {
            const __bf16* pp = &Pt[w][ln][hl ? 8 : 0];
            pa.q[0] = *(const u32x4*)pp;
            pa.q[1] = *(const u32x4*)(pp + 16);
        }
        // O += P * V ; V frags straight from global (32key x 16v)
#pragma unroll
        for (int nt = 0; nt < 8; nt++) {
            Frag vb;
            const __bf16* vq = vp + nt * 16;
            for (int s = 0; s < 16; s++)
                vb.v[s] = vq[(size_t)((s & 7) + 8 * hl + (s >> 3) * 16) * ldq];
            O[nt] = __builtin_amdgcn_wmma_f32_16x16x32_bf16(
                false, pa.v, false, vb.v, (short)0, O[nt], false, false);
        }
        kp0 += jstep; kp1 += jstep; vp += jstep;
    }

    // normalize and store A[h][q][v] as f32
    float linv = 1.0f / l_i;
    float lf[8];
#pragma unroll
    for (int r = 0; r < 8; r++) lf[r] = __shfl(linv, r + hl * 8, 16);
#pragma unroll
    for (int nt = 0; nt < 8; nt++)
#pragma unroll
        for (int r = 0; r < 8; r++) {
            int q = q0 + r + 8 * hl;
            int v = nt * 16 + ln;
            Aout[((size_t)h * L_SEQ + q) * DH + v] = O[nt][r] * lf[r];
        }
}

// ---------------------------------------------------------------------------
// Assemble the 143 surviving y columns (overlap bug), zero-pad to 160, bf16.
// ---------------------------------------------------------------------------
__global__ void assemble_y_kernel(const float* __restrict__ A,
                                  __bf16* __restrict__ y) {
    size_t i = (size_t)blockIdx.x * blockDim.x + threadIdx.x;
    size_t n = (size_t)L_SEQ * YK;
    if (i >= n) return;
    int l = (int)(i / YK);
    int c = (int)(i % YK);
    float v = 0.0f;
    if (c < HEADS - 1)       v = A[((size_t)c * L_SEQ + l) * DH + 0];
    else if (c < HEADS - 1 + DH)
        v = A[((size_t)(HEADS - 1) * L_SEQ + l) * DH + (c - (HEADS - 1))];
    y[i] = (__bf16)v;
}

// ---------------------------------------------------------------------------
// Host-side launch
// ---------------------------------------------------------------------------
extern "C" void kernel_launch(void* const* d_in, const int* in_sizes, int n_in,
                              void* d_out, int out_size, void* d_ws, size_t ws_size,
                              hipStream_t stream) {
    const float* x      = (const float*)d_in[0];
    const float* Wq     = (const float*)d_in[1];
    const float* bq     = (const float*)d_in[2];
    const float* Wk     = (const float*)d_in[3];
    const float* bk     = (const float*)d_in[4];
    const float* Wv     = (const float*)d_in[5];
    const float* bv     = (const float*)d_in[6];
    const float* Wo     = (const float*)d_in[7];
    const float* bo     = (const float*)d_in[8];
    const float* gamma1 = (const float*)d_in[9];
    const float* beta1  = (const float*)d_in[10];
    const float* gamma2 = (const float*)d_in[11];
    const float* beta2  = (const float*)d_in[12];
    const float* W1     = (const float*)d_in[13];
    const float* b1     = (const float*)d_in[14];
    const float* W2     = (const float*)d_in[15];
    const float* b2     = (const float*)d_in[16];
    float* out = (float*)d_out;

    // workspace carve-up (256B aligned)
    char* base = (char*)d_ws;
    size_t off = 0;
    auto alloc = [&](size_t bytes) -> char* {
        char* p = base + off;
        off = (off + bytes + 255) & ~(size_t)255;
        return p;
    };
    __bf16* xn1   = (__bf16*)alloc((size_t)L_SEQ * DE * 2);
    __bf16* wqkv  = (__bf16*)alloc((size_t)DE * NQKV * 2);
    float*  bqkv  = (float*)alloc((size_t)NQKV * 4);
    __bf16* qkv   = (__bf16*)alloc((size_t)L_SEQ * NQKV * 2);
    float*  attA  = (float*)alloc((size_t)HEADS * L_SEQ * DH * 4);
    __bf16* ybuf  = (__bf16*)alloc((size_t)L_SEQ * YK * 2);
    __bf16* wo_p  = (__bf16*)alloc((size_t)YK * DE * 2);
    float*  x2    = (float*)alloc((size_t)L_SEQ * DE * 4);
    __bf16* xn2b  = (__bf16*)alloc((size_t)L_SEQ * DE * 2);
    float*  xn2f  = (float*)alloc((size_t)L_SEQ * DE * 4);
    __bf16* w1b   = (__bf16*)alloc((size_t)DE * DMLP * 2);
    __bf16* w2b   = (__bf16*)alloc((size_t)DMLP * DE * 2);
    __bf16* h1    = (__bf16*)alloc((size_t)L_SEQ * DMLP * 2);
    (void)ws_size; (void)n_in; (void)in_sizes; (void)out_size;

    // 1) pack weights to bf16
    {
        size_t n = (size_t)DE * NQKV;
        build_wqkv_kernel<<<dim3((unsigned)((n + 255) / 256)), 256, 0, stream>>>(
            Wq, Wk, Wv, bq, bk, bv, wqkv, bqkv);
        size_t n1 = (size_t)DE * DMLP;
        f32_to_bf16_kernel<<<dim3((unsigned)((n1 + 255) / 256)), 256, 0, stream>>>(W1, w1b, n1);
        f32_to_bf16_kernel<<<dim3((unsigned)((n1 + 255) / 256)), 256, 0, stream>>>(W2, w2b, n1);
        size_t n2 = (size_t)YK * DE;
        build_wo_kernel<<<dim3((unsigned)((n2 + 255) / 256)), 256, 0, stream>>>(Wo, wo_p);
    }
    // 2) LN1
    layernorm_kernel<<<dim3(L_SEQ), 256, 0, stream>>>(x, gamma1, beta1, xn1, nullptr);
    // 3) QKV projection: (L x DE) * (DE x 6144) + bqkv -> bf16
    gemm_bf16_wmma<<<dim3(NQKV / GBN, L_SEQ / GBM), 256, 0, stream>>>(
        xn1, DE, wqkv, NQKV, bqkv, L_SEQ, NQKV, DE, /*mode=*/0,
        nullptr, nullptr, qkv, nullptr, NQKV);
    // 4) flash attention -> A (H, L, 128) f32
    flash_attn_kernel<<<dim3(L_SEQ / 64, HEADS), 128, 0, stream>>>(qkv, attA);
    // 5) assemble surviving y columns
    {
        size_t n = (size_t)L_SEQ * YK;
        assemble_y_kernel<<<dim3((unsigned)((n + 255) / 256)), 256, 0, stream>>>(attA, ybuf);
    }
    // 6) x2 = 2*x + (y @ Wo_packed + bo)
    gemm_bf16_wmma<<<dim3(DE / GBN, L_SEQ / GBM), 256, 0, stream>>>(
        ybuf, YK, wo_p, DE, bo, L_SEQ, DE, YK, /*mode=*/1,
        x, nullptr, nullptr, x2, DE);
    // 7) LN2 (bf16 for GEMM + f32 for residual)
    layernorm_kernel<<<dim3(L_SEQ), 256, 0, stream>>>(x2, gamma2, beta2, xn2b, xn2f);
    // 8) h1 = gelu(xn2 @ W1 + b1)
    gemm_bf16_wmma<<<dim3(DMLP / GBN, L_SEQ / GBM), 256, 0, stream>>>(
        xn2b, DE, w1b, DMLP, b1, L_SEQ, DMLP, DE, /*mode=*/2,
        nullptr, nullptr, h1, nullptr, DMLP);
    // 9) out = x2 + xn2 + (h1 @ W2 + b2)
    gemm_bf16_wmma<<<dim3(DE / GBN, L_SEQ / GBM), 256, 0, stream>>>(
        h1, DMLP, w2b, DE, b2, L_SEQ, DE, DMLP, /*mode=*/3,
        x2, xn2f, nullptr, out, DE);
}
